// FullMatchLay_54717883351464
// MI455X (gfx1250) — matrix operand order repaired and checked by reading, not compiled, not run on hardware
//
#include <hip/hip_runtime.h>
#include <math.h>

typedef float  v2f  __attribute__((ext_vector_type(2)));
typedef float  v4f  __attribute__((ext_vector_type(4)));
typedef float  v8f  __attribute__((ext_vector_type(8)));
typedef __bf16 v16bf __attribute__((ext_vector_type(16)));

constexpr int kD   = 300;   // = 9*32 + 12 : 9 bf16 K-chunks + f32 tail
constexpr int kKC  = 288;   // bf16-covered K range
constexpr int kTail = 12;   // f32 16x16x4 tail (3 K-steps of 4)
constexpr int kMP  = 20;    // padded to 32 rows -> 2 N-tiles
constexpr int kL   = 2048;
constexpr int kB   = 64;
constexpr int kRowsPerBlock = 256;   // 8 waves * 2 M-tiles * 16 rows
constexpr float kEps = 1e-8f;

#define WMMA_BF16(A, Bm, C) \
    __builtin_amdgcn_wmma_f32_16x16x32_bf16(false, (A), false, (Bm), (short)0, (C), false, false)
#define WMMA_F32(A, Bm, C) \
    __builtin_amdgcn_wmma_f32_16x16x4_f32(false, (A), false, (Bm), (short)0, (C), false, false)

// Build bf16 A fragments (hi, lo, hi-of-square) for one 16x32 tile.
// p points at this lane's row, already offset by kh*16; 16 contiguous floats.
__device__ __forceinline__ void build_frags(const float* __restrict__ p,
                                            v16bf& ah, v16bf& al, v16bf& a2h) {
    float F[16];
    *(v4f*)&F[0]  = __builtin_nontemporal_load(reinterpret_cast<const v4f*>(p));
    *(v4f*)&F[4]  = __builtin_nontemporal_load(reinterpret_cast<const v4f*>(p + 4));
    *(v4f*)&F[8]  = __builtin_nontemporal_load(reinterpret_cast<const v4f*>(p + 8));
    *(v4f*)&F[12] = __builtin_nontemporal_load(reinterpret_cast<const v4f*>(p + 12));
#pragma unroll
    for (int i = 0; i < 16; ++i) {
        float x = F[i];
        __bf16 h = (__bf16)x;          // RNE (v_cvt_pk_bf16_f32)
        ah[i]  = h;
        al[i]  = (__bf16)(x - (float)h);   // exact residual (v_fma_mix_f32_bf16), then RNE
        a2h[i] = (__bf16)(x * x);
    }
}

__global__ __launch_bounds__(256) void fullmatch_wmma_bf16split(
    const float* __restrict__ cont,    // [B, L, D]
    const float* __restrict__ other,   // [B, D]
    const float* __restrict__ weight,  // [MP, D]
    float* __restrict__ out)           // [B, L, MP]
{
    __shared__ __bf16 sUh[32 * kKC];   // hi(other*w^2), rows 20..31 zero
    __shared__ __bf16 sUl[32 * kKC];   // lo(other*w^2)
    __shared__ __bf16 sWh[32 * kKC];   // hi(w^2)
    __shared__ float  sUf[32 * kTail]; // f32 tail of other*w^2 (d = 288..299)
    __shared__ float  sWf[32 * kTail]; // f32 tail of w^2
    __shared__ float  sNb[32];
    __shared__ float  sPart[160];

    const int b    = blockIdx.y;
    const int tid  = threadIdx.x;
    const int lane = tid & 31;
    const int wv   = tid >> 5;

    const float* __restrict__ otherB = other + (size_t)b * kD;

    // ---- Prologue: build split-bf16 B operands in LDS ----
    for (int i = tid; i < 32 * kKC; i += 256) {
        int m = i / kKC;
        int d = i - m * kKC;
        float w  = (m < kMP) ? weight[m * kD + d] : 0.0f;
        float w2 = w * w;
        float u  = w2 * otherB[d];
        __bf16 uh = (__bf16)u;
        sWh[i] = (__bf16)w2;
        sUh[i] = uh;
        sUl[i] = (__bf16)(u - (float)uh);
    }
    for (int i = tid; i < 32 * kTail; i += 256) {
        int m = i / kTail;
        int d = kKC + (i - m * kTail);
        float w  = (m < kMP) ? weight[m * kD + d] : 0.0f;
        float w2 = w * w;
        sWf[i] = w2;
        sUf[i] = w2 * otherB[d];
    }
    // nb[m] = sqrt(sum_d other^2 * w^2): 8 lanes per m.
    if (tid < 160) {
        int m = tid >> 3, j = tid & 7;
        const float* wrow = weight + m * kD;
        float s = 0.0f;
        for (int d = j; d < kD; d += 8) {
            float o = otherB[d];
            float w = wrow[d];
            s = fmaf(o * o, w * w, s);
        }
        sPart[tid] = s;
    }
    __syncthreads();
    if (tid < 32) {
        float s = 0.0f;
        if (tid < kMP) {
            const float* p = &sPart[tid * 8];
            for (int j = 0; j < 8; ++j) s += p[j];
        }
        sNb[tid] = sqrtf(s);
    }
    __syncthreads();

    // ---- Main loop ----
    const int kh = lane >> 4;   // lane half (selects K sub-ranges per ISA layouts)
    const int ln = lane & 15;   // M for A, N for B/C/D
    const int row_base = blockIdx.x * kRowsPerBlock + wv * 32;

    // bf16 A: lane's 16 contiguous floats start at kb + kh*16.
    const float* __restrict__ aA = cont + ((size_t)b * kL + row_base + ln) * kD + kh * 16;
    const float* __restrict__ aB = aA + 16 * kD;
    // bf16 B: 16 consecutive bf16 at column m, K base kb + kh*16.
    const __bf16* __restrict__ uhp = &sUh[ln * kKC + kh * 16];
    const __bf16* __restrict__ ulp = &sUl[ln * kKC + kh * 16];
    const __bf16* __restrict__ whp = &sWh[ln * kKC + kh * 16];

    v8f dA0 = {}, dA1 = {}, nA0 = {}, nA1 = {};
    v8f dB0 = {}, dB1 = {}, nB0 = {}, nB1 = {};

    // unroll 2: double-buffers A/B fragment registers across iterations,
    // removing most WMMA->VALU WAR hazard v_nops.
#pragma unroll 2
    for (int kb = 0; kb < kKC; kb += 32) {
        v16bf Uh0 = *(const v16bf*)(uhp + kb);
        v16bf Uh1 = *(const v16bf*)(uhp + 16 * kKC + kb);
        v16bf Ul0 = *(const v16bf*)(ulp + kb);
        v16bf Ul1 = *(const v16bf*)(ulp + 16 * kKC + kb);
        v16bf Wh0 = *(const v16bf*)(whp + kb);
        v16bf Wh1 = *(const v16bf*)(whp + 16 * kKC + kb);

        v16bf ahA, alA, a2A, ahB, alB, a2B;
        build_frags(aA + kb, ahA, alA, a2A);
        build_frags(aB + kb, ahB, alB, a2B);

        // dot: ah*Uh + ah*Ul + al*Uh  (drops only O(2^-18) al*Ul)
        dA0 = WMMA_BF16(ahA, Uh0, dA0);  dA0 = WMMA_BF16(ahA, Ul0, dA0);  dA0 = WMMA_BF16(alA, Uh0, dA0);
        dA1 = WMMA_BF16(ahA, Uh1, dA1);  dA1 = WMMA_BF16(ahA, Ul1, dA1);  dA1 = WMMA_BF16(alA, Uh1, dA1);
        dB0 = WMMA_BF16(ahB, Uh0, dB0);  dB0 = WMMA_BF16(ahB, Ul0, dB0);  dB0 = WMMA_BF16(alB, Uh0, dB0);
        dB1 = WMMA_BF16(ahB, Uh1, dB1);  dB1 = WMMA_BF16(ahB, Ul1, dB1);  dB1 = WMMA_BF16(alB, Uh1, dB1);
        // na^2: all-positive sum, RNE errors cancel -> single hi term
        nA0 = WMMA_BF16(a2A, Wh0, nA0);  nA1 = WMMA_BF16(a2A, Wh1, nA1);
        nB0 = WMMA_BF16(a2B, Wh0, nB0);  nB1 = WMMA_BF16(a2B, Wh1, nB1);
    }

    // ---- f32 16x16x4 tail for K = 288..299 (exact) ----
    {
        const float* __restrict__ afA = cont + ((size_t)b * kL + row_base + ln) * kD + kh * 2;
        const float* __restrict__ afB = afA + 16 * kD;
        const float* __restrict__ ufp = &sUf[ln * kTail + kh * 2];
        const float* __restrict__ wfp = &sWf[ln * kTail + kh * 2];
#pragma unroll
        for (int t = 0; t < kTail; t += 4) {
            v2f a  = __builtin_nontemporal_load(reinterpret_cast<const v2f*>(afA + kKC + t));
            v2f a2 = a * a;
            v2f c  = __builtin_nontemporal_load(reinterpret_cast<const v2f*>(afB + kKC + t));
            v2f c2 = c * c;
            v2f u0 = *(const v2f*)(ufp + t);
            v2f u1 = *(const v2f*)(ufp + 16 * kTail + t);
            v2f w0 = *(const v2f*)(wfp + t);
            v2f w1 = *(const v2f*)(wfp + 16 * kTail + t);
            dA0 = WMMA_F32(a,  u0, dA0);  dA1 = WMMA_F32(a,  u1, dA1);
            nA0 = WMMA_F32(a2, w0, nA0);  nA1 = WMMA_F32(a2, w1, nA1);
            dB0 = WMMA_F32(c,  u0, dB0);  dB1 = WMMA_F32(c,  u1, dB1);
            nB0 = WMMA_F32(c2, w0, nB0);  nB1 = WMMA_F32(c2, w1, nB1);
        }
    }

    // ---- Epilogue ----
    const float nb0 = sNb[ln];
    const float nb1 = sNb[16 + ln];
    float* __restrict__ obase = out + ((size_t)b * kL + row_base + kh * 8) * kMP;

#pragma unroll
    for (int i = 0; i < 8; ++i) {
        float* rowA = obase + (size_t)i * kMP;
        float* rowB = rowA + 16 * kMP;
        float denA0 = fmaxf(sqrtf(nA0[i]) * nb0, kEps);
        float denB0 = fmaxf(sqrtf(nB0[i]) * nb0, kEps);
        __builtin_nontemporal_store(dA0[i] / denA0, rowA + ln);
        __builtin_nontemporal_store(dB0[i] / denB0, rowB + ln);
        if (ln < kMP - 16) {   // only m = 16..19 valid in N-tile 1
            float denA1 = fmaxf(sqrtf(nA1[i]) * nb1, kEps);
            float denB1 = fmaxf(sqrtf(nB1[i]) * nb1, kEps);
            __builtin_nontemporal_store(dA1[i] / denA1, rowA + 16 + ln);
            __builtin_nontemporal_store(dB1[i] / denB1, rowB + 16 + ln);
        }
    }
}

extern "C" void kernel_launch(void* const* d_in, const int* in_sizes, int n_in,
                              void* d_out, int out_size, void* d_ws, size_t ws_size,
                              hipStream_t stream) {
    (void)in_sizes; (void)n_in; (void)out_size; (void)d_ws; (void)ws_size;
    const float* cont   = (const float*)d_in[0];
    const float* other  = (const float*)d_in[1];
    const float* weight = (const float*)d_in[2];
    float* out = (float*)d_out;

    dim3 grid(kL / kRowsPerBlock, kB);   // 8 x 64 = 512 blocks
    fullmatch_wmma_bf16split<<<grid, 256, 0, stream>>>(cont, other, weight, out);
}
